// SpatioTemporalGCN_24704651887209
// MI455X (gfx1250) — matrix operand order repaired and checked by reading, not compiled, not run on hardware
//
#include <hip/hip_runtime.h>

// ---------------------------------------------------------------------------
// SpatioTemporalGCN for MI455X (gfx1250, wave32, WMMA).
// - Wd@Wp folded (adjacency mix commutes with pointwise linear)
// - BN folded to per-channel scale/shift
// - all GEMMs: bf16 WMMA (v_wmma_f32_16x16x32_bf16), f32 accumulate
// - A-tile LDS staging via async global->LDS (ASYNCcnt) when available
// ---------------------------------------------------------------------------

typedef __bf16 v16bf __attribute__((ext_vector_type(16)));
typedef float  v8f   __attribute__((ext_vector_type(8)));
typedef int    v4i   __attribute__((ext_vector_type(4)));

#if __has_builtin(__builtin_amdgcn_global_load_async_to_lds_b128)
#define HAS_ASYNC_LDS 1
typedef __attribute__((address_space(1))) v4i gv4i_t;  // global 16B chunk
typedef __attribute__((address_space(3))) v4i lv4i_t;  // LDS 16B chunk
#else
#define HAS_ASYNC_LDS 0
#endif

__device__ __forceinline__ void wait_async0() {
#if HAS_ASYNC_LDS
#if __has_builtin(__builtin_amdgcn_s_wait_asynccnt)
  __builtin_amdgcn_s_wait_asynccnt(0);
#else
  asm volatile("s_wait_asynccnt 0x0" ::: "memory");
#endif
#endif
}

__device__ __forceinline__ unsigned short f2bf(float f) {
  union { float f; unsigned u; } a; a.f = f;
  unsigned r = a.u + 0x7FFFu + ((a.u >> 16) & 1u);   // round-to-nearest-even
  return (unsigned short)(r >> 16);
}
__device__ __forceinline__ float bf2f(unsigned short h) {
  union { unsigned u; float f; } a; a.u = ((unsigned)h) << 16;
  return a.f;
}

#define GBM 128
#define GBN 64
#define GBK 32
#define A_STRIDE 40   // ushorts per LDS row (80B, 16B multiple)
#define B_STRIDE 48   // ushorts per LDS row, B stored [col][k] (96B)

union Frag { v16bf v; uint4 q[2]; };
union Acc  { v8f v; float f[8]; };

// C[M,N] = act( (A[M,K] @ W[K,N]) * scale + shift )
// A: bf16(ushort) row-major. W: bf16 row-major. Out: fp32 or bf16.
__global__ __launch_bounds__(256) void gemm_wmma_kernel(
    const unsigned short* __restrict__ Ab,
    const unsigned short* __restrict__ W,
    void* __restrict__ Out, int outBF,
    int M, int N, int K,
    const float* __restrict__ scale, const float* __restrict__ shift, int act) {
  __shared__ alignas(16) unsigned short As[GBM * A_STRIDE];
  __shared__ alignas(16) unsigned short Bs[GBN * B_STRIDE];

  const int tid  = threadIdx.x;
  const int lane = tid & 31;
  const int wid  = tid >> 5;
  const int waveM = wid & 3;   // 4 waves along M
  const int waveN = wid >> 2;  // 2 waves along N
  const int l16 = lane >> 4;
  const int lr  = lane & 15;

  const int bn = blockIdx.x * GBN;
  const int bm = blockIdx.y * GBM;

  Acc acc[2][2] = {};

  for (int k0 = 0; k0 < K; k0 += GBK) {
    // ---- stage A tile [128 x 32] bf16 -> LDS ------------------------------
    // 512 chunks of 16B (8 bf16); 2 per thread. Async global->LDS if present.
#pragma unroll
    for (int i = 0; i < 2; ++i) {
      int idx = tid + i * 256;
      int r = idx >> 2, cv = idx & 3;
      const unsigned short* src = Ab + (size_t)(bm + r) * K + k0 + cv * 8;
      unsigned short* dst = As + r * A_STRIDE + cv * 8;
#if HAS_ASYNC_LDS
      __builtin_amdgcn_global_load_async_to_lds_b128(
          (gv4i_t*)src, (lv4i_t*)dst, 0, 0);
#else
      *(uint4*)dst = *(const uint4*)src;
#endif
    }
    // ---- stage B tile [32 x 64] transposed -> Bs[col][k] ------------------
#pragma unroll
    for (int i = 0; i < 2; ++i) {
      int idx = tid + i * 256;            // 512 chunks of 4 bf16
      int r = idx >> 4, cv = idx & 15;
      uint2 d = *(const uint2*)(W + (size_t)(k0 + r) * N + bn + cv * 4);
      int cb = cv * 4;
      Bs[(cb + 0) * B_STRIDE + r] = (unsigned short)(d.x & 0xffff);
      Bs[(cb + 1) * B_STRIDE + r] = (unsigned short)(d.x >> 16);
      Bs[(cb + 2) * B_STRIDE + r] = (unsigned short)(d.y & 0xffff);
      Bs[(cb + 3) * B_STRIDE + r] = (unsigned short)(d.y >> 16);
    }
    if (k0 + GBK < K) {  // prefetch next K-tile (global_prefetch_b8)
      __builtin_prefetch((const void*)(Ab + (size_t)(bm + (tid >> 3)) * K + k0 + GBK + (tid & 7) * 4), 0, 1);
      __builtin_prefetch((const void*)(W + (size_t)(k0 + GBK + (tid >> 4)) * N + bn), 0, 1);
    }
    wait_async0();
    __syncthreads();

    // ---- fragments per ISA layout -----------------------------------------
    Frag afr[2], bfr[2];
#pragma unroll
    for (int t = 0; t < 2; ++t) {
      // A 16x32: lane holds row lr; e0..7 at k=8*l16.. , e8..15 at k=16+8*l16..
      const unsigned short* pa = As + (waveM * 32 + t * 16 + lr) * A_STRIDE + l16 * 8;
      afr[t].q[0] = *(const uint4*)(pa);
      afr[t].q[1] = *(const uint4*)(pa + 16);
      // B 32x16: lane holds column lr; e0..7 at k=8*l16.. , e8..15 at k=16+8*l16..
      const unsigned short* pb = Bs + (waveN * 32 + t * 16 + lr) * B_STRIDE + l16 * 8;
      bfr[t].q[0] = *(const uint4*)(pb);
      bfr[t].q[1] = *(const uint4*)(pb + 16);
    }
#pragma unroll
    for (int tm = 0; tm < 2; ++tm)
#pragma unroll
      for (int tn = 0; tn < 2; ++tn)
        acc[tm][tn].v = __builtin_amdgcn_wmma_f32_16x16x32_bf16(
            false, afr[tm].v, false, bfr[tn].v, (short)0, acc[tm][tn].v, false, false);
    __syncthreads();
  }

  // ---- epilogue: scale/shift/act, fp32 or bf16 store ----------------------
  float* Of = (float*)Out;
  unsigned short* Ob = (unsigned short*)Out;
#pragma unroll
  for (int tn = 0; tn < 2; ++tn) {
    int ng = bn + waveN * 32 + tn * 16 + lr;
    float sc = scale ? scale[ng] : 1.0f;
    float sh = shift ? shift[ng] : 0.0f;
#pragma unroll
    for (int tm = 0; tm < 2; ++tm) {
      int rbase = bm + waveM * 32 + tm * 16 + 8 * l16;
#pragma unroll
      for (int v = 0; v < 8; ++v) {
        float f = acc[tm][tn].f[v];
        if (scale) f *= sc;
        f += sh;
        if (act == 1) f = f > 0.f ? f : 0.f;
        else if (act == 2) f = 1.f / (1.f + __expf(-f));
        size_t o = (size_t)(rbase + v) * N + ng;
        if (outBF) Ob[o] = f2bf(f); else Of[o] = f;
      }
    }
  }
}

// out[bs,n,c] = epi( sum_m A[n,m] * Z[bs,m,c] ); one block per (b,s).
__global__ __launch_bounds__(256) void amix_kernel(
    const float* __restrict__ Amat, const float* __restrict__ Z,
    void* __restrict__ Out, int C,
    const float* __restrict__ scale, const float* __restrict__ shift,
    int relu, int outBF) {
  __shared__ float sA[24 * 24];
  __shared__ float tile[24 * 512];
  const int bs = blockIdx.x, tid = threadIdx.x;
  for (int i = tid; i < 576; i += 256) sA[i] = Amat[i];
  const int total = 24 * C;
  const float* zb = Z + (size_t)bs * total;
  for (int i = tid; i < total; i += 256) tile[i] = zb[i];
  __syncthreads();
  float* of = (float*)Out;
  unsigned short* ob = (unsigned short*)Out;
  for (int i = tid; i < total; i += 256) {
    int n = i / C, c = i - n * C;
    float acc = 0.f;
#pragma unroll
    for (int m = 0; m < 24; ++m) acc += sA[n * 24 + m] * tile[m * C + c];
    if (scale) acc *= scale[c];
    if (shift) acc += shift[c];
    if (relu) acc = acc > 0.f ? acc : 0.f;
    size_t o = (size_t)bs * total + i;
    if (outBF) ob[o] = f2bf(acc); else of[o] = acc;
  }
}

__global__ void pool_kernel(const float* __restrict__ h3, unsigned short* __restrict__ pooled) {
  int bs = blockIdx.x;
  for (int c = threadIdx.x; c < 512; c += blockDim.x) {
    float acc = 0.f;
    for (int n = 0; n < 24; ++n) acc += h3[((size_t)bs * 24 + n) * 512 + c];
    pooled[(size_t)bs * 512 + c] = f2bf(acc * (1.0f / 24.0f));
  }
}

__global__ void gate_kernel(const float* __restrict__ h3, const float* __restrict__ ca,
                            unsigned short* __restrict__ hg, int total) {
  int idx = blockIdx.x * 256 + threadIdx.x;
  if (idx >= total) return;
  int c = idx & 511;
  size_t m = (size_t)idx >> 9;
  size_t bs = m / 24;
  hg[idx] = f2bf(h3[idx] * ca[bs * 512 + c]);
}

__global__ void dots3_kernel(const unsigned short* __restrict__ sa2,
                             const float* __restrict__ ws3, const float* __restrict__ bs3,
                             float* __restrict__ sa, int M) {
  int m = blockIdx.x * 256 + threadIdx.x;
  if (m >= M) return;
  float acc = bs3[0];
  const unsigned short* r = sa2 + (size_t)m * 64;
#pragma unroll 8
  for (int j = 0; j < 64; ++j) acc += bf2f(r[j]) * ws3[j];
  sa[m] = 1.f / (1.f + __expf(-acc));
}

__global__ void final_kernel(const float* __restrict__ h3, const float* __restrict__ ca,
                             const float* __restrict__ sa, const float* __restrict__ r3,
                             const float* __restrict__ s4, const float* __restrict__ t4,
                             float* __restrict__ out, int total) {
  int idx = blockIdx.x * 256 + threadIdx.x;
  if (idx >= total) return;
  int c = idx & 511;
  size_t m = (size_t)idx >> 9;
  size_t bs = m / 24;
  float f = h3[idx] * ca[bs * 512 + c] * sa[m];
  f = f * s4[c] + t4[c] + r3[idx];
  out[idx] = f > 0.f ? f : 0.f;
}

__global__ void fold_bn_kernel(const float* g, const float* b, const float* m, const float* v,
                               const float* bias, float* s_out, float* c_out, int n) {
  int i = blockIdx.x * 256 + threadIdx.x;
  if (i >= n) return;
  float s = g[i] * rsqrtf(v[i] + 1e-5f);
  float c = b[i] - m[i] * s;
  if (bias) c += bias[i] * s;
  s_out[i] = s;
  c_out[i] = c;
}

__global__ void cvt_kernel(const float* __restrict__ src, unsigned short* __restrict__ dst, int n) {
  int i = blockIdx.x * 256 + threadIdx.x;
  if (i < n) dst[i] = f2bf(src[i]);
}

// Wfused[k,n] = sum_j Wd[k,j] * Wp[j,n]  (bf16 out).  Wd: [K,K], Wp: [K,N].
__global__ void fusew_kernel(const float* __restrict__ Wd, const float* __restrict__ Wp,
                             unsigned short* __restrict__ out, int K, int N) {
  int idx = blockIdx.x * 256 + threadIdx.x;
  if (idx >= K * N) return;
  int k = idx / N, n = idx - k * N;
  float acc = 0.f;
  for (int j = 0; j < K; ++j) acc += Wd[k * K + j] * Wp[j * N + n];
  out[idx] = f2bf(acc);
}

// ---------------------------------------------------------------------------

extern "C" void kernel_launch(void* const* d_in, const int* in_sizes, int n_in,
                              void* d_out, int out_size, void* d_ws, size_t ws_size,
                              hipStream_t stream) {
  (void)in_sizes; (void)n_in; (void)out_size; (void)ws_size;
  const float* x   = (const float*)d_in[0];
  const float* A   = (const float*)d_in[1];
  const float* Wd1 = (const float*)d_in[2];
  const float* Wd2 = (const float*)d_in[3];
  const float* Wd3 = (const float*)d_in[4];
  const float* Wp1 = (const float*)d_in[5];
  const float* bp1 = (const float*)d_in[6];
  const float* Wp2 = (const float*)d_in[7];
  const float* bp2 = (const float*)d_in[8];
  const float* Wp3 = (const float*)d_in[9];
  const float* bp3 = (const float*)d_in[10];
  const float* g2 = (const float*)d_in[11], *b2 = (const float*)d_in[12],
             * m2 = (const float*)d_in[13], *v2 = (const float*)d_in[14];
  const float* g3 = (const float*)d_in[15], *b3 = (const float*)d_in[16],
             * m3 = (const float*)d_in[17], *v3 = (const float*)d_in[18];
  const float* g4 = (const float*)d_in[19], *b4 = (const float*)d_in[20],
             * m4 = (const float*)d_in[21], *v4 = (const float*)d_in[22];
  const float* Wc1 = (const float*)d_in[23], *bc1 = (const float*)d_in[24];
  const float* Wc2 = (const float*)d_in[25], *bc2 = (const float*)d_in[26];
  const float* Wc3 = (const float*)d_in[27], *bc3 = (const float*)d_in[28];
  const float* Ws1 = (const float*)d_in[29], *bs1 = (const float*)d_in[30];
  const float* Ws2 = (const float*)d_in[31], *bs2 = (const float*)d_in[32];
  const float* Ws3 = (const float*)d_in[33], *bs3 = (const float*)d_in[34];
  const float* Wr1 = (const float*)d_in[35], *br1 = (const float*)d_in[36];
  const float* Wr2 = (const float*)d_in[37], *br2 = (const float*)d_in[38];
  const float* Wr3 = (const float*)d_in[39], *br3 = (const float*)d_in[40];
  const float* gr = (const float*)d_in[41], *brn = (const float*)d_in[42],
             * mr = (const float*)d_in[43], *vr = (const float*)d_in[44];

  const size_t M = 49152;  // B*S*N = 8*256*24
  size_t off = 0;
  char* ws = (char*)d_ws;
  auto alloc = [&](size_t bytes) -> void* {
    off = (off + 255) & ~(size_t)255;
    void* p = ws + off;
    off += bytes;
    return p;
  };

  float* z3 = (float*)alloc(M * 512 * 4);  // -> h3 (A-mix3 in-place fp32)
  float* r3 = (float*)alloc(M * 512 * 4);
  char*  R3 = (char*)alloc(M * 512 * 2);   // z1 / h2 / sa chain / pooled / ca1/ca2, h1 high
  char*  R4 = (char*)alloc(M * 512 * 2);   // z2 / hg / r1+r2
  float* ca = (float*)alloc(2048 * 512 * 4);
  unsigned short* xbf = (unsigned short*)alloc(M * 256 * 2);  // bf16 copy of x

  unsigned short* W1bf = (unsigned short*)alloc(256 * 128 * 2);
  unsigned short* W2bf = (unsigned short*)alloc(128 * 256 * 2);
  unsigned short* W3bf = (unsigned short*)alloc(256 * 512 * 2);
  unsigned short* Wr1bf = (unsigned short*)alloc(256 * 64 * 2);
  unsigned short* Wr2bf = (unsigned short*)alloc(64 * 128 * 2);
  unsigned short* Wr3bf = (unsigned short*)alloc(128 * 512 * 2);
  unsigned short* Wc1bf = (unsigned short*)alloc(512 * 128 * 2);
  unsigned short* Wc2bf = (unsigned short*)alloc(128 * 256 * 2);
  unsigned short* Wc3bf = (unsigned short*)alloc(256 * 512 * 2);
  unsigned short* Ws1bf = (unsigned short*)alloc(512 * 128 * 2);
  unsigned short* Ws2bf = (unsigned short*)alloc(128 * 64 * 2);
  float* s2 = (float*)alloc(128 * 4); float* c2 = (float*)alloc(128 * 4);
  float* s3 = (float*)alloc(256 * 4); float* c3 = (float*)alloc(256 * 4);
  float* s4 = (float*)alloc(512 * 4); float* t4 = (float*)alloc(512 * 4);
  float* sR = (float*)alloc(512 * 4); float* cR = (float*)alloc(512 * 4);

  // Region aliases (lifetimes verified: writer always runs after last reader).
  float* z1 = (float*)R3;                                   // M*128 f32
  unsigned short* h1 = (unsigned short*)(R3 + 26214400);    // M*128 bf16
  unsigned short* h2 = (unsigned short*)R3;                 // M*256 bf16 (after z1 dead)
  unsigned short* sa1 = (unsigned short*)R3;                // M*128 bf16 (after h2 dead)
  unsigned short* sa2 = (unsigned short*)(R3 + 13107200);   // M*64 bf16
  float* saV = (float*)(R3 + 20971520);                     // M f32
  unsigned short* pooled = (unsigned short*)(R3 + 22020096);// 2048*512 bf16
  unsigned short* ca1 = (unsigned short*)(R3 + 25165824);   // 2048*128 bf16
  unsigned short* ca2 = (unsigned short*)(R3 + 26214400);   // 2048*256 bf16 (after h1 dead)
  float* z2 = (float*)R4;                                   // M*256 f32
  unsigned short* hg = (unsigned short*)R4;                 // M*512 bf16 (after z2 dead)
  unsigned short* r1 = (unsigned short*)R4;                 // M*64 bf16 (after hg dead)
  unsigned short* r2 = (unsigned short*)(R4 + 8388608);     // M*128 bf16

  float* out = (float*)d_out;
  const int total = (int)(M * 512);

  // ---- prep: BN folds, weight conversions, Wd@Wp fusion, x->bf16 ----
  fold_bn_kernel<<<1, 256, 0, stream>>>(g2, b2, m2, v2, bp1, s2, c2, 128);
  fold_bn_kernel<<<1, 256, 0, stream>>>(g3, b3, m3, v3, bp2, s3, c3, 256);
  fold_bn_kernel<<<2, 256, 0, stream>>>(g4, b4, m4, v4, nullptr, s4, t4, 512);
  fold_bn_kernel<<<2, 256, 0, stream>>>(gr, brn, mr, vr, br3, sR, cR, 512);
  cvt_kernel<<<64, 256, 0, stream>>>(Wr1, Wr1bf, 256 * 64);
  cvt_kernel<<<32, 256, 0, stream>>>(Wr2, Wr2bf, 64 * 128);
  cvt_kernel<<<256, 256, 0, stream>>>(Wr3, Wr3bf, 128 * 512);
  cvt_kernel<<<256, 256, 0, stream>>>(Wc1, Wc1bf, 512 * 128);
  cvt_kernel<<<128, 256, 0, stream>>>(Wc2, Wc2bf, 128 * 256);
  cvt_kernel<<<512, 256, 0, stream>>>(Wc3, Wc3bf, 256 * 512);
  cvt_kernel<<<256, 256, 0, stream>>>(Ws1, Ws1bf, 512 * 128);
  cvt_kernel<<<32, 256, 0, stream>>>(Ws2, Ws2bf, 128 * 64);
  cvt_kernel<<<(int)(M * 256 / 256), 256, 0, stream>>>(x, xbf, (int)(M * 256));
  fusew_kernel<<<128, 256, 0, stream>>>(Wd1, Wp1, W1bf, 256, 128);
  fusew_kernel<<<128, 256, 0, stream>>>(Wd2, Wp2, W2bf, 128, 256);
  fusew_kernel<<<512, 256, 0, stream>>>(Wd3, Wp3, W3bf, 256, 512);

  // ---- GCN blocks: GEMM (WMMA) -> A-mix + BN/bias epilogue ----
  gemm_wmma_kernel<<<dim3(128 / GBN, M / GBM), 256, 0, stream>>>(
      xbf, W1bf, z1, 0, (int)M, 128, 256, nullptr, nullptr, 0);
  amix_kernel<<<2048, 256, 0, stream>>>(A, z1, h1, 128, s2, c2, 1, 1);
  gemm_wmma_kernel<<<dim3(256 / GBN, M / GBM), 256, 0, stream>>>(
      h1, W2bf, z2, 0, (int)M, 256, 128, nullptr, nullptr, 0);
  amix_kernel<<<2048, 256, 0, stream>>>(A, z2, h2, 256, s3, c3, 1, 1);
  gemm_wmma_kernel<<<dim3(512 / GBN, M / GBM), 256, 0, stream>>>(
      h2, W3bf, z3, 0, (int)M, 512, 256, nullptr, nullptr, 0);
  amix_kernel<<<2048, 256, 0, stream>>>(A, z3, z3, 512, nullptr, bp3, 0, 0);  // -> h3 in-place
  float* h3 = z3;

  // ---- channel attention ----
  pool_kernel<<<2048, 256, 0, stream>>>(h3, pooled);
  gemm_wmma_kernel<<<dim3(128 / GBN, 2048 / GBM), 256, 0, stream>>>(
      pooled, Wc1bf, ca1, 1, 2048, 128, 512, nullptr, bc1, 1);
  gemm_wmma_kernel<<<dim3(256 / GBN, 2048 / GBM), 256, 0, stream>>>(
      ca1, Wc2bf, ca2, 1, 2048, 256, 128, nullptr, bc2, 1);
  gemm_wmma_kernel<<<dim3(512 / GBN, 2048 / GBM), 256, 0, stream>>>(
      ca2, Wc3bf, ca, 0, 2048, 512, 256, nullptr, bc3, 2);

  // ---- spatial attention on gated activations ----
  gate_kernel<<<total / 256, 256, 0, stream>>>(h3, ca, hg, total);
  gemm_wmma_kernel<<<dim3(128 / GBN, M / GBM), 256, 0, stream>>>(
      hg, Ws1bf, sa1, 1, (int)M, 128, 512, nullptr, bs1, 1);
  gemm_wmma_kernel<<<dim3(64 / GBN, M / GBM), 256, 0, stream>>>(
      sa1, Ws2bf, sa2, 1, (int)M, 64, 128, nullptr, bs2, 1);
  dots3_kernel<<<(int)(M / 256), 256, 0, stream>>>(sa2, Ws3, bs3, saV, (int)M);

  // ---- residual branch (runs late so r1/r2 reuse the hg region) ----
  gemm_wmma_kernel<<<dim3(64 / GBN, M / GBM), 256, 0, stream>>>(
      xbf, Wr1bf, r1, 1, (int)M, 64, 256, nullptr, br1, 1);
  gemm_wmma_kernel<<<dim3(128 / GBN, M / GBM), 256, 0, stream>>>(
      r1, Wr2bf, r2, 1, (int)M, 128, 64, nullptr, br2, 1);
  gemm_wmma_kernel<<<dim3(512 / GBN, M / GBM), 256, 0, stream>>>(
      r2, Wr3bf, r3, 0, (int)M, 512, 128, sR, cR, 0);

  // ---- final: BN4 + gates + residual + relu ----
  final_kernel<<<total / 256, 256, 0, stream>>>(h3, ca, saV, r3, s4, t4, out, total);
}
